// MultiHeadAttention_13039520710965
// MI455X (gfx1250) — matrix operand order repaired and checked by reading, not compiled, not run on hardware
//
#include <hip/hip_runtime.h>
#include <hip/hip_bf16.h>

// ---------------------------------------------------------------------------
// CDNA5 (gfx1250, wave32) WMMA multi-head attention (transposed-head variant).
//   B=8, S=2048, D=1024, H=16, DH=64
// Pipeline:
//   1) q/k/v = bf16(X @ W^T + b)        -- 3x WMMA GEMM, f32->bf16
//   2) per (b,h): score = q k^T / 8, mask, softmax, ws^T = v attn^T  -- WMMA
//   3) out = ws_scrambled @ Wo^T + bo    -- WMMA GEMM, f32 out
// bf16 tile staging of pre-converted tensors goes through the Tensor Data
// Mover (tensor_load_to_lds + s_wait_tensorcnt) when the builtin exists.
// ---------------------------------------------------------------------------

typedef __attribute__((ext_vector_type(16))) __bf16 v16bf;
typedef __attribute__((ext_vector_type(8)))  float  v8f;
typedef __attribute__((ext_vector_type(4)))  unsigned int u32x4;
typedef __attribute__((ext_vector_type(2)))  unsigned int u32x2;
typedef __attribute__((ext_vector_type(4)))  float        f32x4;
typedef __attribute__((ext_vector_type(8)))  int          i32x8;
typedef __attribute__((ext_vector_type(4)))  int          i32x4;

union FragU  { v16bf v; u32x4 q[2]; };
union Pack8  { u32x4 q; __bf16 h[8]; };
union Pack4  { u32x2 q; __bf16 h[4]; };

#if defined(__HIP_DEVICE_COMPILE__) && __has_builtin(__builtin_amdgcn_tensor_load_to_lds)
#define HAVE_TDM 1
#else
#define HAVE_TDM 0
#endif

#if HAVE_TDM
// Issue one TDM 2-D tile load (bf16 elements) global -> LDS.
// Descriptor packing per CDNA5 ISA:
//   group0: [1:0]=count=1, [63:32]=lds_addr, [120:64]=global_addr, [127:126]=2
//   group1: [17:16]=data_size(1=2B), [20]=pad_en, [24:22]=pad_interval,
//           [31:25]=pad_amount, [79:48]=tensor_dim0, [111:80]=tensor_dim1,
//           [127:112]=tile_dim0, [143:128]=tile_dim1, [207:160]=dim0_stride
// pad: interval code i -> (8<<i) bytes between pads; amount code a -> (a+1) DWORDs.
__device__ __forceinline__ void tdm_load_2d_bf16(
    unsigned lds_off, unsigned long long gaddr, unsigned tdim0, unsigned tdim1,
    unsigned tile0, unsigned tile1, unsigned stride0, unsigned pad_i, unsigned pad_a) {
  u32x4 g0;
  g0[0] = 1u;                                         // count=1 (user mode)
  g0[1] = lds_off;                                    // LDS byte address
  g0[2] = (unsigned)(gaddr & 0xffffffffull);          // global addr [31:0]
  g0[3] = (unsigned)((gaddr >> 32) & 0x1ffffffull) | (2u << 30);  // [56:32]|type=2
  i32x8 g1;
  g1[0] = (int)((1u << 16) | (1u << 20) | (pad_i << 22) | (pad_a << 25));
  g1[1] = (int)((tdim0 & 0xffffu) << 16);             // [63:48] = tensor_dim0 lo
  g1[2] = (int)((tdim0 >> 16) | ((tdim1 & 0xffffu) << 16));
  g1[3] = (int)((tdim1 >> 16) | (tile0 << 16));       // [127:112] = tile_dim0
  g1[4] = (int)tile1;                                 // tile_dim1, tile_dim2=0
  g1[5] = (int)stride0;                               // dim0 stride (elements)
  g1[6] = 0;
  g1[7] = 0;
  i32x4 z4 = {0, 0, 0, 0};
#if __clang_major__ >= 23
  i32x8 z8 = {0, 0, 0, 0, 0, 0, 0, 0};
  __builtin_amdgcn_tensor_load_to_lds(g0, g1, z4, z4, z8, 0);
#else
  __builtin_amdgcn_tensor_load_to_lds(g0, g1, z4, z4, 0);
#endif
}
#endif  // HAVE_TDM

__device__ __forceinline__ v8f wmma_bf16(v16bf a, v16bf b, v8f c) {
  return __builtin_amdgcn_wmma_f32_16x16x32_bf16(false, a, false, b, (short)0, c,
                                                 false, false);
}

// A-fragment (16x32 bf16, M=rows, K=cols) from an LDS region stored row-major.
// Layout (ISA 7.12.2): lane<16 -> M=lane, K 0-7 @ +0, K16-23 @ +32
//                      lane>=16 -> M=lane-16, K 8-15 @ +16, K24-31 @ +48
__device__ __forceinline__ v16bf lds_frag_a(const __bf16* base, int stride_b, int lane) {
  const char* p = (const char*)base + (lane & 15) * stride_b + ((lane & 16) ? 16 : 0);
  FragU u;
  u.q[0] = *(const u32x4*)(p);
  u.q[1] = *(const u32x4*)(p + 32);
  return u.v;
}

// B-fragment (32x16 bf16, K=rows, N=cols): lane -> N=lane%16; element s -> K in
// table order (s<8: 8h+s, s>=8: 16+8h+s-8), h=lane/16.
__device__ __forceinline__ v16bf lds_frag_b(const __bf16* base, int stride_b, int n0,
                                            int lane) {
  const char* p = (const char*)base + (n0 + (lane & 15)) * 2;
  const int hk = (lane & 16) ? 8 : 0;
  v16bf r;
#pragma unroll
  for (int s = 0; s < 16; ++s) {
    int k = (s < 8) ? (hk + s) : (16 + hk + (s - 8));
    r[s] = *(const __bf16*)(p + k * stride_b);
  }
  return r;
}

__device__ __forceinline__ void store_out(float* p, float v)  { *p = v; }
__device__ __forceinline__ void store_out(__bf16* p, float v) { *p = (__bf16)v; }

// ---------------------------------------------------------------------------
// C = A @ W^T + bias.  A: M x K (f32 or bf16), W: N x K f32 (transposed+converted
// to bf16 in LDS), C: M x N (OT).  Block tile 128x128, K-step 32, 8 waves.
// ---------------------------------------------------------------------------
template <bool A_IS_F32, typename OT>
__global__ __launch_bounds__(256) void gemm_bias_kernel(
    const void* __restrict__ Araw, const float* __restrict__ W,
    const float* __restrict__ bias, OT* __restrict__ C, int M, int N, int K) {
  constexpr int BM = 128, BN = 128, BK = 32;
  constexpr int AS = 40;   // LDS A row stride in bf16 (80 B = 64 data + 16 pad)
  constexpr int BS = 136;  // LDS B row stride in bf16 (272 B)
  __shared__ __bf16 As[BM * AS];
  __shared__ __bf16 Bs[BK * BS];

  const int tid = threadIdx.x;
  const int lane = tid & 31;
  const int wave = tid >> 5;
  const int wm = wave >> 1;  // 0..3 -> 32-row band
  const int wn = wave & 1;   // 0..1 -> 64-col band
  const int m0 = blockIdx.y * BM;
  const int n0 = blockIdx.x * BN;

  v8f acc[2][4] = {};

  for (int k0 = 0; k0 < K; k0 += BK) {
    // ---- stage A tile ----
    if constexpr (A_IS_F32) {
      const float* A = (const float*)Araw;
#pragma unroll
      for (int p = 0; p < 4; ++p) {
        int m = p * 32 + (tid >> 3);
        int k4 = (tid & 7) * 4;
        const float* src = &A[(size_t)(m0 + m) * K + k0 + k4];
        f32x4 f = *(const f32x4*)src;
        if (k0 + BK < K) __builtin_prefetch(src + BK, 0, 0);  // next K-tile
        Pack4 pk;
        pk.h[0] = (__bf16)f.x; pk.h[1] = (__bf16)f.y;
        pk.h[2] = (__bf16)f.z; pk.h[3] = (__bf16)f.w;
        *(u32x2*)&As[m * AS + k4] = pk.q;
      }
    } else {
      // bf16 A: let the Tensor Data Mover stage (and pad) the tile.
#if HAVE_TDM
      if (wave == 0)
        tdm_load_2d_bf16((unsigned)(size_t)(void*)As,
                         (unsigned long long)(size_t)((const __bf16*)Araw +
                                                      (size_t)m0 * K + k0),
                         /*tdim0=*/BK, /*tdim1=*/BM, /*tile0=*/BK, /*tile1=*/BM,
                         /*stride0=*/(unsigned)K,
                         /*pad_i=*/3 /*64B*/, /*pad_a=*/3 /*+16B*/);
#else
      const __bf16* A = (const __bf16*)Araw;
#pragma unroll
      for (int p = 0; p < 2; ++p) {
        int m = p * 64 + (tid >> 2);
        int k8 = (tid & 3) * 8;
        *(u32x4*)&As[m * AS + k8] =
            *(const u32x4*)&A[(size_t)(m0 + m) * K + k0 + k8];
      }
#endif
    }
    // ---- stage B tile: Bs[k][n] = bf16(W[n0+n][k0+k])  (transpose) ----
#pragma unroll
    for (int p = 0; p < 4; ++p) {
      int n = p * 32 + (tid >> 3);
      int k4 = (tid & 7) * 4;
      const float* src = &W[(size_t)(n0 + n) * K + k0 + k4];
      f32x4 f = *(const f32x4*)src;
      if (k0 + BK < K) __builtin_prefetch(src + BK, 0, 0);
      Bs[(k4 + 0) * BS + n] = (__bf16)f.x;
      Bs[(k4 + 1) * BS + n] = (__bf16)f.y;
      Bs[(k4 + 2) * BS + n] = (__bf16)f.z;
      Bs[(k4 + 3) * BS + n] = (__bf16)f.w;
    }
#if HAVE_TDM
    if constexpr (!A_IS_F32) {
      if (wave == 0) __builtin_amdgcn_s_wait_tensorcnt(0);
    }
#endif
    __syncthreads();

    v16bf af[2];
#pragma unroll
    for (int i = 0; i < 2; ++i)
      af[i] = lds_frag_a(&As[(wm * 32 + i * 16) * AS], AS * 2, lane);
#pragma unroll
    for (int j = 0; j < 4; ++j) {
      v16bf bf = lds_frag_b(Bs, BS * 2, wn * 64 + j * 16, lane);
#pragma unroll
      for (int i = 0; i < 2; ++i) acc[i][j] = wmma_bf16(af[i], bf, acc[i][j]);
    }
    __syncthreads();
  }

  // ---- epilogue: bias + store.  C/D layout: VGPR r, lane<16 -> (M=r, N=lane);
  //      lane>=16 -> (M=r+8, N=lane-16) ----
#pragma unroll
  for (int i = 0; i < 2; ++i) {
    int row = m0 + wm * 32 + i * 16 + ((lane >> 4) << 3);
#pragma unroll
    for (int j = 0; j < 4; ++j) {
      int col = n0 + wn * 64 + j * 16 + (lane & 15);
      float bv = bias[col];
#pragma unroll
      for (int r = 0; r < 8; ++r)
        store_out(&C[(size_t)(row + r) * N + col], acc[i][j][r] + bv);
    }
  }
}

// ---------------------------------------------------------------------------
// One block per (b,h).  128 threads = 4 waves.
// Phase 1: score[d][e] = sum_s q[d,s] k[e,s]  (64x64, K=2048) via WMMA.
// Phase 2: scale 1/8, mask over e, softmax rows, store attn^T bf16.
// Phase 3: ws^T[s,d] = sum_e v[e,s] attn[d,e]; scatter to scrambled (B,S,D).
// ---------------------------------------------------------------------------
__global__ __launch_bounds__(128) void attention_kernel(
    const __bf16* __restrict__ qp, const __bf16* __restrict__ kp,
    const __bf16* __restrict__ vp, const int* __restrict__ mask,
    __bf16* __restrict__ wsout) {
  constexpr int S = 2048, D = 1024, H = 16, DH = 64;
  constexpr int QS = 136;  // Qs: 64 (d) x 128 (s) bf16, 272 B rows
  constexpr int KS = 72;   // Ks: 128 (s) x 64 (e) bf16, 144 B rows
  constexpr int SS = 68;   // Sc: 64 x 64 f32, 272 B rows
  constexpr int TS = 72;   // At: 64 (e) x 64 (d) bf16, 144 B rows
  __shared__ __bf16 Qs[64 * QS];
  __shared__ __bf16 Ks[128 * KS];
  __shared__ float  Sc[64 * SS];
  __shared__ __bf16 At[64 * TS];

  const int bh = blockIdx.x;
  const int b = bh >> 4;
  const int h = bh & 15;
  const int tid = threadIdx.x;
  const int lane = tid & 31;
  const int wave = tid >> 5;
  const size_t headoff = (size_t)b * S * D + h * DH;

  // ---------------- phase 1: score accumulation over S ----------------
  v8f acc[4] = {};
  for (int sc = 0; sc < S; sc += 128) {
#if HAVE_TDM
    // K-head tile (128 s x 64 e bf16, row stride D) -> Ks, +16B pad per 128B row.
    if (wave == 0)
      tdm_load_2d_bf16((unsigned)(size_t)(void*)Ks,
                       (unsigned long long)(size_t)(kp + headoff + (size_t)sc * D),
                       /*tdim0=*/DH, /*tdim1=*/128, /*tile0=*/DH, /*tile1=*/128,
                       /*stride0=*/D, /*pad_i=*/4 /*128B*/, /*pad_a=*/3 /*+16B*/);
#endif
#pragma unroll
    for (int p = 0; p < 8; ++p) {
      int sl = p * 16 + (tid >> 3);
      int d0 = (tid & 7) * 8;
      Pack8 pq;
      pq.q = *(const u32x4*)&qp[headoff + (size_t)(sc + sl) * D + d0];
#pragma unroll
      for (int i = 0; i < 8; ++i) Qs[(d0 + i) * QS + sl] = pq.h[i];  // transpose
#if !HAVE_TDM
      *(u32x4*)&Ks[sl * KS + d0] =
          *(const u32x4*)&kp[headoff + (size_t)(sc + sl) * D + d0];
#endif
    }
#if HAVE_TDM
    if (wave == 0) __builtin_amdgcn_s_wait_tensorcnt(0);
#endif
    __syncthreads();
#pragma unroll
    for (int ks = 0; ks < 4; ++ks) {
      v16bf a = lds_frag_a(&Qs[(wave * 16) * QS + ks * 32], QS * 2, lane);
#pragma unroll
      for (int j = 0; j < 4; ++j) {
        v16bf bfr = lds_frag_b(&Ks[(ks * 32) * KS], KS * 2, j * 16, lane);
        acc[j] = wmma_bf16(a, bfr, acc[j]);
      }
    }
    __syncthreads();
  }

  // ---------------- phase 2: mask + softmax (rows of 64) ----------------
  {
    int ml = wave * 16 + ((lane >> 4) << 3);
#pragma unroll
    for (int j = 0; j < 4; ++j)
#pragma unroll
      for (int r = 0; r < 8; ++r)
        Sc[(ml + r) * SS + j * 16 + (lane & 15)] = acc[j][r] * 0.125f;
  }
  __syncthreads();
  if (tid < 64) {
    const int d = tid;
    float x[64];
    float mx = -3.0e38f;
#pragma unroll
    for (int e = 0; e < 64; ++e) {
      float v = Sc[d * SS + e];
      v = (mask[b * DH + e] == 0) ? -1.0e10f : v;
      x[e] = v;
      mx = fmaxf(mx, v);
    }
    float sum = 0.f;
#pragma unroll
    for (int e = 0; e < 64; ++e) { x[e] = __expf(x[e] - mx); sum += x[e]; }
    float inv = 1.0f / sum;
#pragma unroll
    for (int e = 0; e < 64; ++e) At[e * TS + d] = (__bf16)(x[e] * inv);  // attn^T
  }
  __syncthreads();

  // ---------------- phase 3: ws^T = v-block @ attn^T ----------------
  v16bf bt[2][4];
#pragma unroll
  for (int ks = 0; ks < 2; ++ks)
#pragma unroll
    for (int j = 0; j < 4; ++j)
      bt[ks][j] = lds_frag_b(&At[(ks * 32) * TS], TS * 2, j * 16, lane);

  for (int st = wave; st < 128; st += 4) {
    const int s0 = st * 16;
    v8f o[4] = {};
#pragma unroll
    for (int ks = 0; ks < 2; ++ks) {
      // A-fragment straight from global: head dim is contiguous in memory.
      const char* p =
          (const char*)(vp + headoff + (size_t)(s0 + (lane & 15)) * D + ks * 32) +
          ((lane & 16) ? 16 : 0);
      FragU u;
      u.q[0] = *(const u32x4*)(p);
      u.q[1] = *(const u32x4*)(p + 32);
#pragma unroll
      for (int j = 0; j < 4; ++j) o[j] = wmma_bf16(u.v, bt[ks][j], o[j]);
    }
    // scatter into reference's scrambled (B, S, D) layout:
    // flat = (d*H + h)*S + s  ->  row = flat>>10, col = flat&1023
#pragma unroll
    for (int j = 0; j < 4; ++j) {
      int d = j * 16 + (lane & 15);
#pragma unroll
      for (int r = 0; r < 8; ++r) {
        int s = s0 + r + ((lane >> 4) << 3);
        int row = ((d * H + h) << 1) + (s >> 10);
        int col = s & 1023;
        wsout[(size_t)b * S * D + (size_t)row * D + col] = (__bf16)o[j][r];
      }
    }
  }
}

// ---------------------------------------------------------------------------
extern "C" void kernel_launch(void* const* d_in, const int* in_sizes, int n_in,
                              void* d_out, int out_size, void* d_ws, size_t ws_size,
                              hipStream_t stream) {
  const float* Q    = (const float*)d_in[0];
  const float* Kin  = (const float*)d_in[1];
  const float* V    = (const float*)d_in[2];
  const int*   mask = (const int*)d_in[3];
  const float* Wq   = (const float*)d_in[4];
  const float* bq   = (const float*)d_in[5];
  const float* Wk   = (const float*)d_in[6];
  const float* bk   = (const float*)d_in[7];
  const float* Wv   = (const float*)d_in[8];
  const float* bv   = (const float*)d_in[9];
  const float* Wo   = (const float*)d_in[10];
  const float* bo   = (const float*)d_in[11];
  float* out = (float*)d_out;

  constexpr int B = 8, S = 2048, D = 1024;
  constexpr size_t TENSOR_B = (size_t)B * S * D * 2;  // 32 MB bf16
  char* ws = (char*)d_ws;
  __bf16* qp  = (__bf16*)(ws);
  __bf16* kp  = (__bf16*)(ws + TENSOR_B);
  __bf16* vp  = (__bf16*)(ws + 2 * TENSOR_B);
  __bf16* wsS = (__bf16*)(ws + 3 * TENSOR_B);

  dim3 blk(256);
  dim3 grid(D / 128, (B * S) / 128);

  hipLaunchKernelGGL((gemm_bias_kernel<true, __bf16>), grid, blk, 0, stream,
                     (const void*)Q, Wq, bq, qp, B * S, D, D);
  hipLaunchKernelGGL((gemm_bias_kernel<true, __bf16>), grid, blk, 0, stream,
                     (const void*)Kin, Wk, bk, kp, B * S, D, D);
  hipLaunchKernelGGL((gemm_bias_kernel<true, __bf16>), grid, blk, 0, stream,
                     (const void*)V, Wv, bv, vp, B * S, D, D);

  hipLaunchKernelGGL(attention_kernel, dim3(128), dim3(128), 0, stream,
                     qp, kp, vp, mask, wsS);

  hipLaunchKernelGGL((gemm_bias_kernel<false, float>), grid, blk, 0, stream,
                     (const void*)wsS, Wo, bo, out, B * S, D, D);
}
